// MultiHeadAttention_69337952027079
// MI455X (gfx1250) — matrix operand order repaired and checked by reading, not compiled, not run on hardware
//
#include <hip/hip_runtime.h>
#include <hip/hip_bf16.h>

#define EMBED 768
#define HEADS 12
#define HDIM  64
#define BATCH 2
#define SEQ   4096
#define MTOT  (BATCH*SEQ)   // 8192 rows
#define NQKV  (3*EMBED)     // 2304

#if defined(__AMDGCN__) && __has_builtin(__builtin_amdgcn_global_load_async_to_lds_b128)
#define HAVE_ASYNC_LDS 1
#else
#define HAVE_ASYNC_LDS 0
#endif

typedef __attribute__((ext_vector_type(16))) __bf16 v16bf;
typedef __attribute__((ext_vector_type(8)))  __bf16 v8bf;
typedef __attribute__((ext_vector_type(8)))  float  v8f;

union ABFrag { v16bf v; v8bf h[2]; };

__device__ __forceinline__ v8f wmma_bf16(v16bf a, v16bf b, v8f c) {
  return __builtin_amdgcn_wmma_f32_16x16x32_bf16(false, a, false, b, (short)0, c,
                                                 false, false);
}

// 16B global->LDS copy: async path on gfx1250, sync fallback elsewhere.
__device__ __forceinline__ void lds_copy16(void* lds_dst, const void* gsrc) {
#if HAVE_ASYNC_LDS
  typedef int v4i __attribute__((vector_size(16)));
  typedef __attribute__((address_space(1))) v4i* gp_t;
  typedef __attribute__((address_space(3))) v4i* lp_t;
  __builtin_amdgcn_global_load_async_to_lds_b128((gp_t)(void*)gsrc, (lp_t)lds_dst, 0, 0);
#else
  *(v8bf*)lds_dst = *(const v8bf*)gsrc;
#endif
}

__device__ __forceinline__ void lds_copy_wait() {
#if HAVE_ASYNC_LDS
#if __has_builtin(__builtin_amdgcn_s_wait_asynccnt)
  __builtin_amdgcn_s_wait_asynccnt(0);
#else
  asm volatile("s_wait_asynccnt 0" ::: "memory");
#endif
#endif
}

// A fragment: 16x32 tile (MxK), row-major, row stride ld elements.
// lanes 0-15: row=lane, elems = K 0..7,16..23 ; lanes 16-31: K 8..15,24..31
__device__ __forceinline__ v16bf load_a_frag(const __bf16* base, int ld, int lane) {
  int row = lane & 15;
  int k0  = (lane & 16) ? 8 : 0;
  const __bf16* p = base + (size_t)row * ld + k0;
  ABFrag u;
  u.h[0] = *(const v8bf*)(p);
  u.h[1] = *(const v8bf*)(p + 16);
  return u.v;
}

// B fragment (32x16 KxN) where B^T is row-major (Bt[col][k] contiguous, stride ld).
// lane=col; lanes 0-15: K 0..7,16..23 ; lanes 16-31: K 8..15,24..31
__device__ __forceinline__ v16bf load_bt_frag(const __bf16* base, int ld, int lane) {
  int col = lane & 15;
  int k0  = (lane & 16) ? 8 : 0;
  const __bf16* p = base + (size_t)col * ld + k0;
  ABFrag u;
  u.h[0] = *(const v8bf*)(p);
  u.h[1] = *(const v8bf*)(p + 16);
  return u.v;
}

__global__ void cvt_f32_bf16(const float* __restrict__ in, __bf16* __restrict__ out, int n) {
  int i = blockIdx.x * blockDim.x + threadIdx.x;
  int stride = gridDim.x * blockDim.x;
  for (; i < n; i += stride) out[i] = (__bf16)in[i];
}

// transpose-convert: in [R][C] f32 -> out [C][R] bf16
__global__ void cvt_t_bf16(const float* __restrict__ in, __bf16* __restrict__ out,
                           int R, int C) {
  int i = blockIdx.x * blockDim.x + threadIdx.x;
  int n = R * C;
  int stride = gridDim.x * blockDim.x;
  for (; i < n; i += stride) {
    int r = i / C, c = i - r * C;
    out[(size_t)c * R + r] = (__bf16)in[i];
  }
}

// ---- QKV projection: [8192,768] x Wt[2304,768] -> Q,K [B,H,T,D] ; V transposed [B,H,D,T]
// double-buffered async LDS staging
__global__ void qkv_gemm(const __bf16* __restrict__ Xb, const __bf16* __restrict__ Wt,
                         __bf16* __restrict__ Qb, __bf16* __restrict__ Kb,
                         __bf16* __restrict__ Vt) {
  __shared__ __align__(16) __bf16 As[2][16 * 32];    // [row][k]
  __shared__ __align__(16) __bf16 Bs[2][256 * 32];   // [col][k]
  int tid = threadIdx.x, lane = tid & 31, wv = tid >> 5;
  int row0 = blockIdx.x * 16;
  int colBlk = blockIdx.y * 256;

  auto stage = [&](int k0, int buf) {
    if (tid < 64) {
      int r = tid >> 2, c8 = tid & 3;
      lds_copy16(&As[buf][tid * 8], Xb + (size_t)(row0 + r) * EMBED + k0 + c8 * 8);
    }
#pragma unroll
    for (int i = 0; i < 8; ++i) {
      int idx = tid + i * 128;                       // 1024 x v8bf
      int r = idx >> 2, c8 = idx & 3;
      lds_copy16(&Bs[buf][idx * 8], Wt + (size_t)(colBlk + r) * EMBED + k0 + c8 * 8);
    }
    if (k0 + 32 < EMBED)
      __builtin_prefetch(Wt + (size_t)(colBlk + (tid << 1)) * EMBED + k0 + 32, 0, 3);
  };

  v8f acc[4] = {};
  stage(0, 0);
  lds_copy_wait();
  __syncthreads();
  for (int it = 0; it < EMBED / 32; ++it) {
    int cur = it & 1;
    if (it + 1 < EMBED / 32) stage((it + 1) * 32, cur ^ 1);   // overlap with compute
    v16bf a = load_a_frag(As[cur], 32, lane);
#pragma unroll
    for (int j = 0; j < 4; ++j) {
      v16bf b = load_bt_frag(Bs[cur] + (wv * 64 + j * 16) * 32, 32, lane);
      acc[j] = wmma_bf16(a, b, acc[j]);
    }
    lds_copy_wait();
    __syncthreads();
  }

  int hi8 = (lane & 16) ? 8 : 0;
#pragma unroll
  for (int j = 0; j < 4; ++j) {
#pragma unroll
    for (int e = 0; e < 8; ++e) {
      int m = row0 + e + hi8;
      int f = colBlk + wv * 64 + j * 16 + (lane & 15);
      int bb = m >> 12, t = m & (SEQ - 1);
      __bf16 val = (__bf16)acc[j][e];
      if (f < EMBED) {                               // Q
        int h = f >> 6, d = f & 63;
        Qb[(((size_t)bb * HEADS + h) * SEQ + t) * HDIM + d] = val;
      } else if (f < 2 * EMBED) {                    // K
        int ec = f - EMBED;
        int h = ec >> 6, d = ec & 63;
        Kb[(((size_t)bb * HEADS + h) * SEQ + t) * HDIM + d] = val;
      } else {                                       // V, transposed [B,H,D,T]
        int ec = f - 2 * EMBED;
        int h = ec >> 6, d = ec & 63;
        Vt[(((size_t)bb * HEADS + h) * HDIM + d) * SEQ + t] = val;
      }
    }
  }
}

// ---- FlashAttention: grid (T/64, B*H), 4 waves x 16 query rows,
// double-buffered K/V async staging
__global__ void flash_attn(const __bf16* __restrict__ Qb, const __bf16* __restrict__ Kb,
                           const __bf16* __restrict__ Vt, __bf16* __restrict__ Ob) {
  __shared__ __align__(16) __bf16 Ks[2][64 * 64];    // [s][d]
  __shared__ __align__(16) __bf16 Vs[2][64 * 64];    // [d][s] (transposed)
  __shared__ __align__(16) __bf16 Ps[4 * 16 * 64];
  int tid = threadIdx.x, lane = tid & 31, wv = tid >> 5;
  int bh = blockIdx.y;
  int bb = bh / HEADS, h = bh - bb * HEADS;
  const __bf16* Q  = Qb + (size_t)bh * SEQ * HDIM;
  const __bf16* K  = Kb + (size_t)bh * SEQ * HDIM;
  const __bf16* Vh = Vt + (size_t)bh * HDIM * SEQ;
  int q0 = blockIdx.x * 64 + wv * 16;

  auto stage = [&](int s0, int buf) {
#pragma unroll
    for (int i = 0; i < 4; ++i) {
      int idx = tid + i * 128;                       // 512 x v8bf per buffer
      int r = idx >> 3, c8 = idx & 7;
      lds_copy16(&Ks[buf][idx * 8], K + (size_t)(s0 + r) * HDIM + c8 * 8);
      lds_copy16(&Vs[buf][idx * 8], Vh + (size_t)r * SEQ + s0 + c8 * 8);
    }
    if (s0 + 64 < SEQ)
      __builtin_prefetch(K + (size_t)(s0 + 64) * HDIM + tid * 32, 0, 3);
  };

  // Q fragments, pre-scaled by 1/sqrt(64)=0.125 (exact power of two in bf16)
  v16bf aq[2];
#pragma unroll
  for (int kc = 0; kc < 2; ++kc) {
    ABFrag u; u.v = load_a_frag(Q + (size_t)q0 * HDIM + kc * 32, HDIM, lane);
#pragma unroll
    for (int e = 0; e < 16; ++e) u.v[e] = (__bf16)((float)u.v[e] * 0.125f);
    aq[kc] = u.v;
  }

  float mrow[8], lrow[8];
#pragma unroll
  for (int e = 0; e < 8; ++e) { mrow[e] = -3.0e38f; lrow[e] = 0.0f; }
  v8f o[4] = {};

  stage(0, 0);
  lds_copy_wait();
  __syncthreads();
  for (int it = 0; it < SEQ / 64; ++it) {
    int cur = it & 1;
    if (it + 1 < SEQ / 64) stage((it + 1) * 64, cur ^ 1);     // overlap with compute

    // S = (Q*scale) * K^T   -> 4 tiles of 16x16 f32
    v8f sc[4];
#pragma unroll
    for (int j = 0; j < 4; ++j) {
      v8f s = {};
#pragma unroll
      for (int kc = 0; kc < 2; ++kc)
        s = wmma_bf16(aq[kc],
                      load_bt_frag(Ks[cur] + (size_t)(j * 16) * HDIM + kc * 32, HDIM, lane), s);
      sc[j] = s;
    }

    // online softmax; per-element-row stats, 16-lane allreduce groups
    float pv[4][8];
#pragma unroll
    for (int e = 0; e < 8; ++e) {
      float cmax = sc[0][e];
#pragma unroll
      for (int j = 1; j < 4; ++j) cmax = fmaxf(cmax, sc[j][e]);
#pragma unroll
      for (int msk = 1; msk < 16; msk <<= 1) cmax = fmaxf(cmax, __shfl_xor(cmax, msk, 16));
      float mnew = fmaxf(mrow[e], cmax);
      float rescale = __expf(mrow[e] - mnew);
      float rsum = 0.0f;
#pragma unroll
      for (int j = 0; j < 4; ++j) {
        float p = __expf(sc[j][e] - mnew);
        pv[j][e] = p;
        rsum += p;
      }
#pragma unroll
      for (int msk = 1; msk < 16; msk <<= 1) rsum += __shfl_xor(rsum, msk, 16);
      lrow[e] = lrow[e] * rescale + rsum;
      mrow[e] = mnew;
#pragma unroll
      for (int j = 0; j < 4; ++j) o[j][e] *= rescale;
    }

    // C-layout -> A-layout via per-wave (private) LDS tile; in-wave LDS RAW
    // is ordered by compiler-inserted s_wait_dscnt, no block barrier needed.
    {
      int hi8 = (lane & 16) ? 8 : 0;
      __bf16* pw = Ps + wv * (16 * 64);
#pragma unroll
      for (int j = 0; j < 4; ++j)
#pragma unroll
        for (int e = 0; e < 8; ++e)
          pw[(size_t)(e + hi8) * 64 + j * 16 + (lane & 15)] = (__bf16)pv[j][e];
    }

    // O += P * V   (V consumed transposed: contiguous B fragments)
    v16bf ap[2];
#pragma unroll
    for (int kc = 0; kc < 2; ++kc)
      ap[kc] = load_a_frag(Ps + wv * (16 * 64) + kc * 32, 64, lane);
#pragma unroll
    for (int j = 0; j < 4; ++j) {
#pragma unroll
      for (int kc = 0; kc < 2; ++kc) {
        v16bf bv = load_bt_frag(Vs[cur] + (size_t)(j * 16) * 64 + kc * 32, 64, lane);
        o[j] = wmma_bf16(ap[kc], bv, o[j]);
      }
    }

    lds_copy_wait();      // next chunk staged
    __syncthreads();      // all waves done with buffers
  }

  int hi8 = (lane & 16) ? 8 : 0;
#pragma unroll
  for (int e = 0; e < 8; ++e) {
    float inv = 1.0f / lrow[e];
    int t = q0 + e + hi8;
#pragma unroll
    for (int j = 0; j < 4; ++j) {
      int d = j * 16 + (lane & 15);
      Ob[((size_t)bb * SEQ + t) * EMBED + h * HDIM + d] = (__bf16)(o[j][e] * inv);
    }
  }
}

// ---- out-projection + fused LayerNorm: one block = 16 full rows of 768
__global__ void outproj_ln(const __bf16* __restrict__ Ab, const __bf16* __restrict__ Wt,
                           const float* __restrict__ bias, const float* __restrict__ g,
                           const float* __restrict__ beta, float* __restrict__ out) {
  __shared__ __align__(16) __bf16 As[16 * 32];       // [row][k]
  __shared__ __align__(16) __bf16 Bs[768 * 32];      // [col][k]  48 KB
  __shared__ float redS[4][16], redQ[4][16], muA[16], rsA[16];
  int tid = threadIdx.x, lane = tid & 31, wv = tid >> 5;
  int row0 = blockIdx.x * 16;
  v8f acc[12] = {};
  for (int k0 = 0; k0 < EMBED; k0 += 32) {
    __syncthreads();
    if (tid < 64) {
      int r = tid >> 2, c8 = tid & 3;
      lds_copy16(&As[tid * 8], Ab + (size_t)(row0 + r) * EMBED + k0 + c8 * 8);
    }
#pragma unroll
    for (int i = 0; i < 24; ++i) {
      int idx = tid + i * 128;                       // 3072 x v8bf
      int r = idx >> 2, c8 = idx & 3;
      lds_copy16(&Bs[idx * 8], Wt + (size_t)r * EMBED + k0 + c8 * 8);
    }
    if (k0 + 32 < EMBED)
      __builtin_prefetch(Wt + (size_t)(tid * 6) * EMBED + k0 + 32, 0, 3);
    lds_copy_wait();
    __syncthreads();
    v16bf a = load_a_frag(As, 32, lane);
#pragma unroll
    for (int j = 0; j < 12; ++j) {
      v16bf b = load_bt_frag(Bs + (wv * 192 + j * 16) * 32, 32, lane);
      acc[j] = wmma_bf16(a, b, acc[j]);
    }
  }
  // bias + LN partial stats (acc becomes y)
  float psum[8], psq[8];
#pragma unroll
  for (int e = 0; e < 8; ++e) { psum[e] = 0.f; psq[e] = 0.f; }
#pragma unroll
  for (int j = 0; j < 12; ++j) {
    int col = wv * 192 + j * 16 + (lane & 15);
    float bcol = bias[col];
#pragma unroll
    for (int e = 0; e < 8; ++e) {
      float v = acc[j][e] + bcol;
      acc[j][e] = v;
      psum[e] += v;
      psq[e]  += v * v;
    }
  }
#pragma unroll
  for (int e = 0; e < 8; ++e) {
#pragma unroll
    for (int msk = 1; msk < 16; msk <<= 1) {
      psum[e] += __shfl_xor(psum[e], msk, 16);
      psq[e]  += __shfl_xor(psq[e],  msk, 16);
    }
  }
  if (lane == 0 || lane == 16) {
    int hi8 = (lane & 16) ? 8 : 0;
#pragma unroll
    for (int e = 0; e < 8; ++e) { redS[wv][e + hi8] = psum[e]; redQ[wv][e + hi8] = psq[e]; }
  }
  __syncthreads();
  if (tid < 16) {
    float s = redS[0][tid] + redS[1][tid] + redS[2][tid] + redS[3][tid];
    float q = redQ[0][tid] + redQ[1][tid] + redQ[2][tid] + redQ[3][tid];
    float mu = s * (1.0f / EMBED);
    float var = q * (1.0f / EMBED) - mu * mu;
    muA[tid] = mu;
    rsA[tid] = rsqrtf(var + 1e-5f);
  }
  __syncthreads();
  int hi8 = (lane & 16) ? 8 : 0;
#pragma unroll
  for (int j = 0; j < 12; ++j) {
    int col = wv * 192 + j * 16 + (lane & 15);
    float gc = g[col], bc = beta[col];
#pragma unroll
    for (int e = 0; e < 8; ++e) {
      int row = e + hi8;
      out[(size_t)(row0 + row) * EMBED + col] =
          (acc[j][e] - muA[row]) * rsA[row] * gc + bc;
    }
  }
}

extern "C" void kernel_launch(void* const* d_in, const int* in_sizes, int n_in,
                              void* d_out, int out_size, void* d_ws, size_t ws_size,
                              hipStream_t stream) {
  (void)in_sizes; (void)n_in; (void)out_size; (void)ws_size;
  const float* x     = (const float*)d_in[0];
  const float* w_qkv = (const float*)d_in[1];
  const float* w_out = (const float*)d_in[2];
  const float* b_out = (const float*)d_in[3];
  const float* ln_g  = (const float*)d_in[4];
  const float* ln_b  = (const float*)d_in[5];
  float* out = (float*)d_out;

  char* ws = (char*)d_ws;
  size_t off = 0;
  auto alloc = [&](size_t bytes) {
    char* p = ws + off;
    off += (bytes + 255) & ~(size_t)255;
    return p;
  };
  __bf16* Xb     = (__bf16*)alloc((size_t)MTOT * EMBED * 2);
  __bf16* Wqkvt  = (__bf16*)alloc((size_t)NQKV * EMBED * 2);   // [2304][768]
  __bf16* Woutt  = (__bf16*)alloc((size_t)EMBED * EMBED * 2);  // [768][768] transposed
  __bf16* Qb     = (__bf16*)alloc((size_t)BATCH * HEADS * SEQ * HDIM * 2);
  __bf16* Kb     = (__bf16*)alloc((size_t)BATCH * HEADS * SEQ * HDIM * 2);
  __bf16* Vt     = (__bf16*)alloc((size_t)BATCH * HEADS * HDIM * SEQ * 2);
  __bf16* Atb    = (__bf16*)alloc((size_t)MTOT * EMBED * 2);

  cvt_f32_bf16<<<512, 256, 0, stream>>>(x, Xb, MTOT * EMBED);
  cvt_t_bf16<<<1024, 256, 0, stream>>>(w_qkv, Wqkvt, EMBED, NQKV);
  cvt_t_bf16<<<512, 256, 0, stream>>>(w_out, Woutt, EMBED, EMBED);

  qkv_gemm<<<dim3(MTOT / 16, NQKV / 256), 128, 0, stream>>>(Xb, Wqkvt, Qb, Kb, Vt);
  flash_attn<<<dim3(SEQ / 64, BATCH * HEADS), 128, 0, stream>>>(Qb, Kb, Vt, Atb);
  outproj_ln<<<MTOT / 16, 128, 0, stream>>>(Atb, Woutt, b_out, ln_g, ln_b, out);
}
